// attention_137438953770
// MI455X (gfx1250) — matrix operand order repaired and checked by reading, not compiled, not run on hardware
//
#include <hip/hip_runtime.h>
#include <hip/hip_fp16.h>

typedef __attribute__((ext_vector_type(16))) _Float16 v16h;
typedef __attribute__((ext_vector_type(8)))  _Float16 v8h;
typedef __attribute__((ext_vector_type(2)))  __fp16   v2fp16;
typedef __attribute__((ext_vector_type(8)))  float    v8f;

#define B_    2
#define N_    2048
#define DIN   1024
#define H_    16
#define DH    64
#define QKV3  3072           // 3*H*DH
#define M_    (B_ * N_)      // 4096 rows total

static __device__ __forceinline__ unsigned pkh2(float a, float b) {
  union { v2fp16 h; unsigned u; } cv;
  cv.h = __builtin_amdgcn_cvt_pkrtz(a, b);    // v_cvt_pk_rtz_f16_f32
  return cv.u;
}

// ---------------------------------------------------------------------------
// fp32 -> fp16 elementwise convert
// ---------------------------------------------------------------------------
__global__ void cvt_f32_to_f16(const float* __restrict__ src,
                               _Float16* __restrict__ dst, int n) {
  int i = blockIdx.x * blockDim.x + threadIdx.x;
  if (i < n) dst[i] = (_Float16)src[i];
}

// w: K x Nn row-major (fp32)  ->  wT: Nn x K row-major (fp16)
__global__ void transpose_cvt(const float* __restrict__ w,
                              _Float16* __restrict__ wT, int K, int Nn) {
  int i = blockIdx.x * blockDim.x + threadIdx.x;
  if (i < K * Nn) {
    int n = i / K, k = i - n * K;
    wT[i] = (_Float16)w[(size_t)k * Nn + n];
  }
}

// vT[(b*H+h)*DH + d][n] = qkv[(b*N+n)*QKV3 + h*192 + 128 + d]
__global__ void build_vT(const _Float16* __restrict__ qkv,
                         _Float16* __restrict__ vT) {
  int i = blockIdx.x * blockDim.x + threadIdx.x;   // B*H*DH*N = 4M
  int n  = i & (N_ - 1);
  int r  = i >> 11;
  int d  = r & (DH - 1);
  int bh = r >> 6;
  int h  = bh & (H_ - 1);
  int b  = bh >> 4;
  vT[i] = qkv[(size_t)(b * N_ + n) * QKV3 + h * 192 + 128 + d];
}

// ---------------------------------------------------------------------------
// WMMA GEMM computing C^T = W(Nmat x K) @ X^T(K x M):
//   A-frag = weight rows (contiguous from WT), B-frag = x rows (contiguous).
//   C^T layout: VGPR g = output column, lane = output row -> packed stores.
// One wave produces a 16-row x 64-col strip of OUT (B-frag reused 4x).
// ---------------------------------------------------------------------------
template <int Nmat, bool OUT16>
__global__ void gemm_wmma(const _Float16* __restrict__ X,   // M x 1024
                          const _Float16* __restrict__ WT,  // Nmat x 1024
                          const float* __restrict__ bias,   // Nmat
                          _Float16* __restrict__ C16,
                          float* __restrict__ C32) {
  constexpr int K = 1024;
  const int lane = threadIdx.x & 31;
  const int wave = threadIdx.x >> 5;
  const int wid  = blockIdx.x * (blockDim.x >> 5) + wave;
  constexpr int ntiles = Nmat / 64;
  const int mt = wid / ntiles;
  const int nt = wid - mt * ntiles;
  const int row0 = mt * 16;      // x rows (output rows)
  const int col0 = nt * 64;      // weight rows (output cols)
  const int c  = lane & 15;
  const int hi = lane >> 4;

  v8f acc[4] = {};
  const _Float16* xRow = X + (size_t)(row0 + c) * K;   // B-frag: lane = x row

  for (int k = 0; k < K; k += 32) {
    v16h xb;                                        // B: 16 contiguous K per half
    *((v8h*)&xb)     = *(const v8h*)(xRow + k + hi * 16);
    *((v8h*)&xb + 1) = *(const v8h*)(xRow + k + hi * 16 + 8);
#pragma unroll
    for (int t = 0; t < 4; ++t) {
      const _Float16* wRow = WT + (size_t)(col0 + t * 16 + c) * K + k;
      v16h wa;                                      // A: row = lane&15, 16 contiguous K per half
      *((v8h*)&wa)     = *(const v8h*)(wRow + hi * 16);
      *((v8h*)&wa + 1) = *(const v8h*)(wRow + hi * 16 + 8);
      acc[t] = __builtin_amdgcn_wmma_f32_16x16x32_f16(
          false, wa, false, xb, (short)0, acc[t], false, false);
    }
  }

#pragma unroll
  for (int t = 0; t < 4; ++t) {
    const int colb = col0 + t * 16 + hi * 8;        // 8 contiguous output cols
    const v8f bv = *(const v8f*)(bias + colb);
    if (OUT16) {
      union { v8h v; unsigned u[4]; } O;
#pragma unroll
      for (int g = 0; g < 4; ++g)
        O.u[g] = pkh2(acc[t][2 * g] + bv[2 * g], acc[t][2 * g + 1] + bv[2 * g + 1]);
      *(v8h*)(C16 + (size_t)(row0 + c) * Nmat + colb) = O.v;      // b128 store
    } else {
      v8f r;
#pragma unroll
      for (int g = 0; g < 8; ++g) r[g] = acc[t][g] + bv[g];
      *(v8f*)(C32 + (size_t)(row0 + c) * Nmat + colb) = r;        // 2x b128
    }
  }
}

// ---------------------------------------------------------------------------
// Flash attention, transposed: S^T = K @ Q^T so softmax reductions are
// in-lane (lane = q column). O^T = V^T @ P^T; P^T built from S^T C-layout
// with a single 4-dword lane-pair exchange. No LDS, no barriers.
// ---------------------------------------------------------------------------
__global__ void flash_attn(const _Float16* __restrict__ qkv,
                           const _Float16* __restrict__ vT,
                           _Float16* __restrict__ outh) {
  const int lane = threadIdx.x & 31;
  const int wave = threadIdx.x >> 5;
  const int wid  = blockIdx.x * (blockDim.x >> 5) + wave;  // 0..4095
  const int qt = wid & 127;
  const int bh = wid >> 7;
  const int h  = bh & 15;
  const int b  = bh >> 4;
  const int c  = lane & 15;
  const int hi = lane >> 4;

  // Q^T B-fragments: lane c = q column, halves = head-dim (contiguous)
  const _Float16* qRow = qkv + (size_t)(b * N_ + qt * 16 + c) * QKV3 + h * 192;
  v16h qb0, qb1;
  *((v8h*)&qb0)     = *(const v8h*)(qRow + hi * 16);
  *((v8h*)&qb0 + 1) = *(const v8h*)(qRow + hi * 16 + 8);
  *((v8h*)&qb1)     = *(const v8h*)(qRow + 32 + hi * 16);
  *((v8h*)&qb1 + 1) = *(const v8h*)(qRow + 32 + hi * 16 + 8);
#pragma unroll
  for (int i = 0; i < 16; ++i) {      // fold softmax scale (exact in f16)
    qb0[i] *= (_Float16)0.125f;
    qb1[i] *= (_Float16)0.125f;
  }

  float m = -1e30f, l = 0.f;          // per-lane: one q column
  v8f o[4] = {};                      // O^T: VGPR g = d, lane = q col

  const _Float16* kBase = qkv + (size_t)(b * N_) * QKV3 + h * 192 + 64;
  const _Float16* vBase = vT + (size_t)(bh * DH) * N_;

  for (int j = 0; j < N_; j += 32) {
    // ---- S^T tiles: A = key rows (contiguous d), B = q columns ----
    v8f s[2];
#pragma unroll
    for (int st = 0; st < 2; ++st) {
      const _Float16* kRow = kBase + (size_t)(j + st * 16 + c) * QKV3;
      v16h ka0, ka1;
      *((v8h*)&ka0)     = *(const v8h*)(kRow + hi * 16);
      *((v8h*)&ka0 + 1) = *(const v8h*)(kRow + hi * 16 + 8);
      *((v8h*)&ka1)     = *(const v8h*)(kRow + 32 + hi * 16);
      *((v8h*)&ka1 + 1) = *(const v8h*)(kRow + 32 + hi * 16 + 8);
      v8f z = {};
      z = __builtin_amdgcn_wmma_f32_16x16x32_f16(false, ka0, false, qb0,
                                                 (short)0, z, false, false);
      z = __builtin_amdgcn_wmma_f32_16x16x32_f16(false, ka1, false, qb1,
                                                 (short)0, z, false, false);
      s[st] = z;
    }

    // ---- online softmax: in-lane reduce + one partner exchange ----
    float tm = fmaxf(s[0][0], s[1][0]);
#pragma unroll
    for (int g = 1; g < 8; ++g) tm = fmaxf(tm, fmaxf(s[0][g], s[1][g]));
    tm = fmaxf(tm, __shfl_xor(tm, 16, 32));
    const float nm = fmaxf(m, tm);
    const float corr = __expf(m - nm);
    m = nm;

    float e0[8], e1[8], own = 0.f;
#pragma unroll
    for (int g = 0; g < 8; ++g) {
      e0[g] = __expf(s[0][g] - m);
      e1[g] = __expf(s[1][g] - m);
      own += e0[g] + e1[g];
    }
    own += __shfl_xor(own, 16, 32);
    l = l * corr + own;
#pragma unroll
    for (int t = 0; t < 4; ++t)
#pragma unroll
      for (int g = 0; g < 8; ++g) o[t][g] *= corr;

    // ---- build P^T B-fragment (32 keys x 16 q) via lane-pair exchange ----
    unsigned pk0[4], pk1[4], ex[4];
#pragma unroll
    for (int d = 0; d < 4; ++d) {
      pk0[d] = pkh2(e0[2 * d], e0[2 * d + 1]);   // own tile-0 rows (g, g+1)
      pk1[d] = pkh2(e1[2 * d], e1[2 * d + 1]);   // own tile-1 rows
    }
#pragma unroll
    for (int d = 0; d < 4; ++d) {
      const unsigned send = hi ? pk0[d] : pk1[d];
      ex[d] = (unsigned)__shfl_xor((int)send, 16, 32);
    }
    union { v16h v; unsigned u[8]; } P;
#pragma unroll
    for (int d = 0; d < 4; ++d) {
      P.u[d]     = hi ? ex[d]  : pk0[d];   // keys (lo half of this lane's K)
      P.u[4 + d] = hi ? pk1[d] : ex[d];    // keys (hi half)
    }

    // ---- O^T += V^T @ P^T (A = vT rows: contiguous keys) ----
#pragma unroll
    for (int t = 0; t < 4; ++t) {
      const _Float16* vRow = vBase + (size_t)(t * 16 + c) * N_ + j;
      v16h va;
      *((v8h*)&va)     = *(const v8h*)(vRow + hi * 16);
      *((v8h*)&va + 1) = *(const v8h*)(vRow + hi * 16 + 8);
      o[t] = __builtin_amdgcn_wmma_f32_16x16x32_f16(
          false, va, false, P.v, (short)0, o[t], false, false);
    }
  }

  // ---- normalize; d-contiguous per lane -> packed b128 stores ----
  const float inv = 1.0f / l;
  _Float16* outp = outh + (size_t)(b * N_ + qt * 16 + c) * DIN + h * DH;
#pragma unroll
  for (int t = 0; t < 4; ++t) {
    union { v8h v; unsigned u[4]; } O;
#pragma unroll
    for (int g = 0; g < 4; ++g)
      O.u[g] = pkh2(o[t][2 * g] * inv, o[t][2 * g + 1] * inv);
    *(v8h*)(outp + t * 16 + hi * 8) = O.v;
  }
}

// ---------------------------------------------------------------------------
extern "C" void kernel_launch(void* const* d_in, const int* in_sizes, int n_in,
                              void* d_out, int out_size, void* d_ws,
                              size_t ws_size, hipStream_t stream) {
  const float* x      = (const float*)d_in[0];
  const float* w_qkv  = (const float*)d_in[1];
  const float* b_qkv  = (const float*)d_in[2];
  const float* w_proj = (const float*)d_in[3];
  const float* b_proj = (const float*)d_in[4];
  float* out = (float*)d_out;

  char* ws = (char*)d_ws;
  _Float16* xh     = (_Float16*)ws; ws += (size_t)M_ * DIN * 2;      //  8 MB
  _Float16* wqkvT  = (_Float16*)ws; ws += (size_t)QKV3 * DIN * 2;    //  6 MB
  _Float16* wprojT = (_Float16*)ws; ws += (size_t)DIN * DIN * 2;     //  2 MB
  _Float16* qkvh   = (_Float16*)ws; ws += (size_t)M_ * QKV3 * 2;     // 24 MB
  _Float16* vTbuf  = (_Float16*)ws; ws += (size_t)B_ * H_ * DH * N_ * 2; // 8 MB
  _Float16* attnh  = (_Float16*)ws; ws += (size_t)M_ * DIN * 2;      //  8 MB

  const int T = 256;
  cvt_f32_to_f16<<<(M_ * DIN + T - 1) / T, T, 0, stream>>>(x, xh, M_ * DIN);
  transpose_cvt<<<(DIN * QKV3 + T - 1) / T, T, 0, stream>>>(w_qkv, wqkvT, DIN, QKV3);
  transpose_cvt<<<(DIN * DIN + T - 1) / T, T, 0, stream>>>(w_proj, wprojT, DIN, DIN);

  // QKV projection: (4096x1024) x (1024x3072) -> f16
  gemm_wmma<QKV3, true><<<(M_ / 16) * (QKV3 / 64) / 4, 128, 0, stream>>>(
      xh, wqkvT, b_qkv, qkvh, nullptr);

  build_vT<<<(B_ * H_ * DH * N_) / T, T, 0, stream>>>(qkvh, vTbuf);

  // Flash attention: 4096 q-tiles, 4 waves/block
  flash_attn<<<(B_ * H_ * (N_ / 16)) / 4, 128, 0, stream>>>(qkvh, vTbuf, attnh);

  // Output projection: (4096x1024) x (1024x1024) -> f32 into d_out
  gemm_wmma<DIN, false><<<(M_ / 16) * (DIN / 64) / 4, 128, 0, stream>>>(
      attnh, wprojT, b_proj, nullptr, out);
}
